// SelfAttention_4200478015740
// MI455X (gfx1250) — compile-verified, matrix operand order checked by reading, not run on hardware
//
#include <hip/hip_runtime.h>
#include <hip/hip_bf16.h>

// Problem constants (match reference)
#define BATCH 4
#define SLEN  2048
#define DMODEL 1024
#define NHEAD 16
#define HDIM  64
#define MROWS (BATCH * SLEN)          // 8192

typedef __bf16 bf16_t;
typedef __attribute__((ext_vector_type(16))) __bf16 v16bf;
typedef __attribute__((ext_vector_type(8)))  __bf16 v8bf;
typedef __attribute__((ext_vector_type(8)))  float  v8f;
typedef __attribute__((ext_vector_type(4)))  int    v4i;

// ---- CDNA5 async global->LDS path (guarded; falls back cleanly) ----
#if defined(__has_builtin)
#if __has_builtin(__builtin_amdgcn_global_load_async_to_lds_b128) && \
    __has_builtin(__builtin_amdgcn_s_wait_asynccnt)
#define HAVE_ASYNC_LDS 1
#endif
#endif

#if defined(HAVE_ASYNC_LDS)
typedef __attribute__((address_space(1))) v4i* glb_v4ip;
typedef __attribute__((address_space(3))) v4i* lds_v4ip;
__device__ __forceinline__ void stage16(bf16_t* l, const bf16_t* g) {
  __builtin_amdgcn_global_load_async_to_lds_b128((glb_v4ip)(g), (lds_v4ip)(l),
                                                 0, 0);
}
// wait until at most N async ops remain in flight (in-order completion per
// wave => previous stage's transfers have retired, current ones keep flying)
template <int N>
__device__ __forceinline__ void stage_wait() {
  __builtin_amdgcn_s_wait_asynccnt(N);
}
#else
__device__ __forceinline__ void stage16(bf16_t* l, const bf16_t* g) {
  *(v8bf*)l = *(const v8bf*)g;
}
template <int N>
__device__ __forceinline__ void stage_wait() {}
#endif

// ---------- helpers ----------

// float -> bf16 round-to-nearest-even
__device__ __forceinline__ bf16_t f2bf(float x) {
  union { float f; unsigned u; } c; c.f = x;
  unsigned r = (c.u + 0x7FFFu + ((c.u >> 16) & 1u)) >> 16;
  unsigned short h = (unsigned short)r;
  return __builtin_bit_cast(bf16_t, h);
}

// Load one 16x32 bf16 WMMA operand fragment from "16 rows x K-contiguous"
// storage (row-major, `stride` elements between rows). Works for A (rows=M)
// and for B given transposed (rows=N, contiguous dim=K). CDNA5 ISA 7.12.2:
//   lanes 0-15 : row=row0+lane,    elems 0..7 -> k 0..7,  8..15 -> k 16..23
//   lanes 16-31: row=row0+lane-16, elems 0..7 -> k 8..15, 8..15 -> k 24..31
__device__ __forceinline__ v16bf load_frag(const bf16_t* base, int stride,
                                           int row0, int lane) {
  int r  = row0 + (lane & 15);
  int kq = (lane >> 4) << 3;                 // 0 or 8
  const bf16_t* p = base + (long)r * stride + kq;
  v8bf lo = *(const v8bf*)(p);
  v8bf hi = *(const v8bf*)(p + 16);
  return __builtin_shufflevector(lo, hi, 0,1,2,3,4,5,6,7,8,9,10,11,12,13,14,15);
}

__device__ __forceinline__ v8f wmma_bf16(v16bf a, v16bf b, v8f c) {
  return __builtin_amdgcn_wmma_f32_16x16x32_bf16(false, a, false, b,
                                                 (short)0, c, false, false);
}

// ---------- conversion kernels ----------

__global__ void __launch_bounds__(256) cvt_bf16(const float* __restrict__ src,
                                                bf16_t* __restrict__ dst, int n) {
  int i = blockIdx.x * 256 + threadIdx.x;
  if (i < n) dst[i] = f2bf(src[i]);
}

// WT[n*K + k] = W[k*N + n]  (convert + transpose so WMMA B-frags are contiguous)
__global__ void __launch_bounds__(256) cvt_bf16_T(const float* __restrict__ W,
                                                  bf16_t* __restrict__ WT,
                                                  int K, int N) {
  int i = blockIdx.x * 256 + threadIdx.x;
  if (i < K * N) {
    int k = i / N, n = i % N;
    WT[(long)n * K + k] = f2bf(W[i]);
  }
}

// ---------- blocked GEMM: 64x128 block tile, 16x64 wave tile ----------
// Double-buffered async LDS staging: next k-slab's global->LDS copies are in
// flight while the current slab feeds the WMMAs.
// C = A[MxK] * BT[NxK]^T
// mode 0: C bf16 row-major [M,N]
// mode 1: C bf16 per-head-transposed V layout [B,H,HDIM,SLEN]
// mode 2: Cf f32 = acc + resid  (row-major [M,N])
__global__ void __launch_bounds__(256) gemm_v2(const bf16_t* __restrict__ A,
                                               const bf16_t* __restrict__ BT,
                                               bf16_t* __restrict__ Cb,
                                               float* __restrict__ Cf,
                                               const float* __restrict__ resid,
                                               int M, int N, int K, int mode) {
  __shared__ __align__(16) bf16_t As[2][64][32];
  __shared__ __align__(16) bf16_t Bs[2][128][32];

  int tid  = threadIdx.x;
  int lane = tid & 31;
  int wave = tid >> 5;
  int tiles_n = N / 128;
  int bm = blockIdx.x / tiles_n;
  int bn = blockIdx.x % tiles_n;

  const bf16_t* Ablk = A  + (long)bm * 64  * K;
  const bf16_t* Bblk = BT + (long)bn * 128 * K;

  int wm = (wave >> 1) * 16;      // 0/16/32/48
  int wn = (wave & 1) * 64;       // 0/64

  // staging coordinates (one b128 for A, two for B, per thread)
  int ar = tid >> 2,  ac = (tid & 3) * 8;     // A: 64 rows x 32 cols
  int br = tid >> 2,  bc = (tid & 3) * 8;     // B: 128 rows x 32 cols

  auto stage_tile = [&](int buf, int k0) {
    stage16(&As[buf][ar][ac],      Ablk + (long)ar * K + k0 + ac);
    stage16(&Bs[buf][br][bc],      Bblk + (long)br * K + k0 + bc);
    stage16(&Bs[buf][br + 64][bc], Bblk + (long)(br + 64) * K + k0 + bc);
  };

  v8f acc[4] = {{}, {}, {}, {}};

  const int nsteps = K >> 5;
  stage_tile(0, 0);                      // prologue
  for (int i = 0; i < nsteps; i++) {
    int cur = i & 1;
    if (i + 1 < nsteps) {
      stage_tile(cur ^ 1, (i + 1) << 5); // overlap next slab with compute
      __builtin_prefetch(Ablk + (long)ar * K + ((i + 2) << 5) + ac, 0, 3);
      __builtin_prefetch(Bblk + (long)br * K + ((i + 2) << 5) + bc, 0, 3);
      stage_wait<3>();                   // only previous 3 must have retired
    } else {
      stage_wait<0>();
    }
    __syncthreads();                     // slab `cur` visible to all waves

    v16bf a  = load_frag(&As[cur][wm][0],      32, 0, lane);
    v16bf b0 = load_frag(&Bs[cur][wn +  0][0], 32, 0, lane);
    v16bf b1 = load_frag(&Bs[cur][wn + 16][0], 32, 0, lane);
    v16bf b2 = load_frag(&Bs[cur][wn + 32][0], 32, 0, lane);
    v16bf b3 = load_frag(&Bs[cur][wn + 48][0], 32, 0, lane);
    acc[0] = wmma_bf16(a, b0, acc[0]);
    acc[1] = wmma_bf16(a, b1, acc[1]);
    acc[2] = wmma_bf16(a, b2, acc[2]);
    acc[3] = wmma_bf16(a, b3, acc[3]);

    __syncthreads();                     // readers done before slab reuse
  }

  int half = lane >> 4;
#pragma unroll
  for (int sub = 0; sub < 4; sub++) {
    int col = bn * 128 + wn + sub * 16 + (lane & 15);
#pragma unroll
    for (int i = 0; i < 8; i++) {
      int row = bm * 64 + wm + half * 8 + i;
      if (mode == 0) {
        Cb[(long)row * N + col] = f2bf(acc[sub][i]);
      } else if (mode == 1) {            // [B,H,HDIM,SLEN]
        int b = row >> 11, s = row & (SLEN - 1);
        int h = col >> 6,  d = col & (HDIM - 1);
        Cb[(((long)((b << 4) + h) * HDIM) + d) * SLEN + s] = f2bf(acc[sub][i]);
      } else {
        long idx = (long)row * N + col;
        Cf[idx] = acc[sub][i] + resid[idx];
      }
    }
  }
}

// ---------- Flash attention ----------
// grid: (SLEN/64, BATCH*NHEAD), block 128 (4 waves). Wave owns 16 q rows.
// K/V chunks double-buffered in LDS via async copies, shared by the 4 waves.
// Qh, Kh: [B,S,H*HDIM] bf16; VhT: [B,H,HDIM,S] bf16; ctx: [B,S,H*HDIM] bf16.
__global__ void __launch_bounds__(128) flash_attn(const bf16_t* __restrict__ Qh,
                                                  const bf16_t* __restrict__ Kh,
                                                  const bf16_t* __restrict__ VhT,
                                                  bf16_t* __restrict__ ctx) {
  __shared__ __align__(16) bf16_t Ks[2][32][64];     // [key][d]
  __shared__ __align__(16) bf16_t Vs[2][64][32];     // [d][key]
  __shared__ __align__(16) bf16_t p_lds[4][16][32];  // per-wave P relayout

  int tid  = threadIdx.x;
  int lane = tid & 31;
  int w    = tid >> 5;
  int bh   = blockIdx.y;                 // b*NHEAD + h
  int b    = bh >> 4, h = bh & 15;
  int qbase = blockIdx.x * 64 + w * 16;

  const bf16_t* qp = Qh  + (long)b * SLEN * DMODEL + h * HDIM;
  const bf16_t* kp = Kh  + (long)b * SLEN * DMODEL + h * HDIM;
  const bf16_t* vp = VhT + (long)bh * HDIM * SLEN;

  // Q tile (16x64) as two A-fragments, resident across the whole K sweep
  v16bf qa0 = load_frag(qp + 0,  DMODEL, qbase, lane);
  v16bf qa1 = load_frag(qp + 32, DMODEL, qbase, lane);

  v8f o0 = {}, o1 = {}, o2 = {}, o3 = {};
  float m[8], l[8];
#pragma unroll
  for (int i = 0; i < 8; i++) { m[i] = -3.0e38f; l[i] = 0.f; }

  // staging coords: K chunk 32x64, V chunk 64x32; 128 threads x 4 b128 total
  int kr0 = tid >> 3,          kc0 = (tid & 7) * 8;        // rows 0..15
  int kr1 = (tid >> 3) + 16;                                // rows 16..31
  int vr0 = tid >> 2,          vc0 = (tid & 3) * 8;        // rows 0..31
  int vr1 = (tid >> 2) + 32;                                // rows 32..63

  auto stage_kv = [&](int buf, int kb) {
    stage16(&Ks[buf][kr0][kc0], kp + (long)(kb + kr0) * DMODEL + kc0);
    stage16(&Ks[buf][kr1][kc0], kp + (long)(kb + kr1) * DMODEL + kc0);
    stage16(&Vs[buf][vr0][vc0], vp + (long)vr0 * SLEN + kb + vc0);
    stage16(&Vs[buf][vr1][vc0], vp + (long)vr1 * SLEN + kb + vc0);
  };

  const float scale = 0.125f;            // 1/sqrt(64)
  const int nch = SLEN / 32;             // 64 key chunks

  stage_kv(0, 0);                        // prologue
  for (int i = 0; i < nch; i++) {
    int kb  = i << 5;
    int cur = i & 1;
    if (i + 1 < nch) {
      stage_kv(cur ^ 1, kb + 32);        // overlap next chunk with compute
      stage_wait<4>();
    } else {
      stage_wait<0>();
    }
    __syncthreads();

    // scores S[16q x 32keys] = Q(16x64) . K^T : 4 WMMAs from LDS
    v8f s0 = {}, s1 = {};
    {
      v16bf k00 = load_frag(&Ks[cur][0][0] + 0,  64, 0,  lane);
      v16bf k01 = load_frag(&Ks[cur][0][0] + 32, 64, 0,  lane);
      s0 = wmma_bf16(qa0, k00, s0);
      s0 = wmma_bf16(qa1, k01, s0);
      v16bf k10 = load_frag(&Ks[cur][0][0] + 0,  64, 16, lane);
      v16bf k11 = load_frag(&Ks[cur][0][0] + 32, 64, 16, lane);
      s1 = wmma_bf16(qa0, k10, s1);
      s1 = wmma_bf16(qa1, k11, s1);
    }

    float p0[8], p1[8], alpha[8];
#pragma unroll
    for (int i2 = 0; i2 < 8; i2++) {
      float a0 = s0[i2] * scale;
      float a1 = s1[i2] * scale;
      // row max across the 16 lanes of this half-wave (row lives in one half)
      float cm = fmaxf(a0, a1);
      cm = fmaxf(cm, __shfl_xor(cm, 1));
      cm = fmaxf(cm, __shfl_xor(cm, 2));
      cm = fmaxf(cm, __shfl_xor(cm, 4));
      cm = fmaxf(cm, __shfl_xor(cm, 8));
      float mn = fmaxf(m[i2], cm);
      alpha[i2] = __expf(m[i2] - mn);
      m[i2] = mn;
      p0[i2] = __expf(a0 - mn);
      p1[i2] = __expf(a1 - mn);
      float rs = p0[i2] + p1[i2];
      rs += __shfl_xor(rs, 1);
      rs += __shfl_xor(rs, 2);
      rs += __shfl_xor(rs, 4);
      rs += __shfl_xor(rs, 8);
      l[i2] = l[i2] * alpha[i2] + rs;
    }
#pragma unroll
    for (int i2 = 0; i2 < 8; i2++) {
      o0[i2] *= alpha[i2]; o1[i2] *= alpha[i2];
      o2[i2] *= alpha[i2]; o3[i2] *= alpha[i2];
    }

    // relayout P (C-frag: column-per-lane) -> A-frag (row-per-lane) via LDS
    int colp = lane & 15, halfp = lane >> 4;
#pragma unroll
    for (int i2 = 0; i2 < 8; i2++) {
      p_lds[w][halfp * 8 + i2][colp]      = f2bf(p0[i2]);
      p_lds[w][halfp * 8 + i2][colp + 16] = f2bf(p1[i2]);
    }
    v16bf pa = load_frag(&p_lds[w][0][0], 32, 0, lane);

    // O += P(16x32) * V_chunk(32x64): 4 WMMAs, B-frags from Vs[d][key]
    v16bf vb;
    vb = load_frag(&Vs[cur][ 0][0], 32, 0, lane); o0 = wmma_bf16(pa, vb, o0);
    vb = load_frag(&Vs[cur][16][0], 32, 0, lane); o1 = wmma_bf16(pa, vb, o1);
    vb = load_frag(&Vs[cur][32][0], 32, 0, lane); o2 = wmma_bf16(pa, vb, o2);
    vb = load_frag(&Vs[cur][48][0], 32, 0, lane); o3 = wmma_bf16(pa, vb, o3);

    __syncthreads();                     // readers done before buffer reuse
  }

#pragma unroll
  for (int i = 0; i < 8; i++) {
    float inv = 1.0f / l[i];
    o0[i] *= inv; o1[i] *= inv; o2[i] *= inv; o3[i] *= inv;
  }

  int col = lane & 15, half = lane >> 4;
#pragma unroll
  for (int i = 0; i < 8; i++) {
    int s = qbase + half * 8 + i;
    long ro = ((long)b * SLEN + s) * DMODEL + h * HDIM;
    ctx[ro +  0 + col] = f2bf(o0[i]);
    ctx[ro + 16 + col] = f2bf(o1[i]);
    ctx[ro + 32 + col] = f2bf(o2[i]);
    ctx[ro + 48 + col] = f2bf(o3[i]);
  }
}

// ---------- LayerNorm (one block per row) ----------
__global__ void __launch_bounds__(256) ln_kernel(const float* __restrict__ x,
                                                 const float* __restrict__ gamma,
                                                 const float* __restrict__ beta,
                                                 float* __restrict__ out) {
  __shared__ float ssum[8], ssq[8];
  long row = blockIdx.x;
  const float* xr = x + row * DMODEL;
  float s = 0.f, s2 = 0.f;
  for (int i = threadIdx.x; i < DMODEL; i += 256) {
    float v = xr[i]; s += v; s2 += v * v;
  }
#pragma unroll
  for (int msk = 1; msk < 32; msk <<= 1) {
    s  += __shfl_xor(s,  msk);
    s2 += __shfl_xor(s2, msk);
  }
  if ((threadIdx.x & 31) == 0) { ssum[threadIdx.x >> 5] = s; ssq[threadIdx.x >> 5] = s2; }
  __syncthreads();
  float ts = 0.f, ts2 = 0.f;
#pragma unroll
  for (int i = 0; i < 8; i++) { ts += ssum[i]; ts2 += ssq[i]; }
  float mean = ts * (1.0f / DMODEL);
  float var  = ts2 * (1.0f / DMODEL) - mean * mean;
  float rstd = rsqrtf(var + 1e-6f);
  for (int i = threadIdx.x; i < DMODEL; i += 256) {
    float v = (xr[i] - mean) * rstd;
    out[row * DMODEL + i] = v * gamma[i] + beta[i];
  }
}

// ---------- launch ----------
extern "C" void kernel_launch(void* const* d_in, const int* in_sizes, int n_in,
                              void* d_out, int out_size, void* d_ws, size_t ws_size,
                              hipStream_t stream) {
  const float* q     = (const float*)d_in[0];
  const float* k     = (const float*)d_in[1];
  const float* v     = (const float*)d_in[2];
  const float* Wq    = (const float*)d_in[3];
  const float* Wk    = (const float*)d_in[4];
  const float* Wv    = (const float*)d_in[5];
  const float* Wo    = (const float*)d_in[6];
  const float* gamma = (const float*)d_in[7];
  const float* beta  = (const float*)d_in[8];
  float* out = (float*)d_out;

  const long NX = (long)MROWS * DMODEL;          // 8,388,608 elements
  const long NW = (long)DMODEL * DMODEL;         // 1,048,576 elements
  char* ws = (char*)d_ws;

  // workspace layout (total ~109 MB; regions reused once producers retire)
  bf16_t* qb  = (bf16_t*)(ws + 0);               // 16.8 MB
  bf16_t* kb  = (bf16_t*)(ws + NX * 2);          // 16.8 MB
  bf16_t* vb  = (bf16_t*)(ws + NX * 4);          // 16.8 MB
  bf16_t* WqT = (bf16_t*)(ws + NX * 6);          // 2 MB each
  bf16_t* WkT = (bf16_t*)(ws + NX * 6 + NW * 2);
  bf16_t* WvT = (bf16_t*)(ws + NX * 6 + NW * 4);
  bf16_t* WoT = (bf16_t*)(ws + NX * 6 + NW * 6);
  bf16_t* Qh  = (bf16_t*)(ws + NX * 6 + NW * 8);
  bf16_t* Kh  = (bf16_t*)(ws + NX * 8 + NW * 8);
  bf16_t* VhT = (bf16_t*)(ws + NX * 10 + NW * 8);
  bf16_t* ctx   = qb;                            // reuse: qb dead after proj
  float*  preln = (float*)(ws + NX * 2);         // reuse kb+vb region (33.6 MB)

  // 1) convert activations and (transposed) weights to bf16
  cvt_bf16<<<(int)(NX / 256), 256, 0, stream>>>(q, qb, (int)NX);
  cvt_bf16<<<(int)(NX / 256), 256, 0, stream>>>(k, kb, (int)NX);
  cvt_bf16<<<(int)(NX / 256), 256, 0, stream>>>(v, vb, (int)NX);
  cvt_bf16_T<<<(int)(NW / 256), 256, 0, stream>>>(Wq, WqT, DMODEL, DMODEL);
  cvt_bf16_T<<<(int)(NW / 256), 256, 0, stream>>>(Wk, WkT, DMODEL, DMODEL);
  cvt_bf16_T<<<(int)(NW / 256), 256, 0, stream>>>(Wv, WvT, DMODEL, DMODEL);
  cvt_bf16_T<<<(int)(NW / 256), 256, 0, stream>>>(Wo, WoT, DMODEL, DMODEL);

  // 2) QKV projections (blocked WMMA GEMM, double-buffered async staging)
  int gblocks = (MROWS / 64) * (DMODEL / 128);   // 1024
  gemm_v2<<<gblocks, 256, 0, stream>>>(qb, WqT, Qh,  nullptr, nullptr,
                                       MROWS, DMODEL, DMODEL, 0);
  gemm_v2<<<gblocks, 256, 0, stream>>>(kb, WkT, Kh,  nullptr, nullptr,
                                       MROWS, DMODEL, DMODEL, 0);
  gemm_v2<<<gblocks, 256, 0, stream>>>(vb, WvT, VhT, nullptr, nullptr,
                                       MROWS, DMODEL, DMODEL, 1);

  // 3) flash attention (WMMA QK^T + online softmax + WMMA PV,
  //    double-buffered async K/V staging)
  dim3 agrid(SLEN / 64, BATCH * NHEAD);
  flash_attn<<<agrid, 128, 0, stream>>>(Qh, Kh, VhT, ctx);

  // 4) output projection + residual (WMMA, f32 out)
  gemm_v2<<<gblocks, 256, 0, stream>>>(ctx, WoT, nullptr, preln, q,
                                       MROWS, DMODEL, DMODEL, 2);

  // 5) LayerNorm
  ln_kernel<<<MROWS, 256, 0, stream>>>(preln, gamma, beta, out);
}
